// NetVLAD_19430432047599
// MI455X (gfx1250) — matrix operand order repaired and checked by reading, not compiled, not run on hardware
//
#include <hip/hip_runtime.h>

// Problem constants (from reference): x[64,128,56,56], conv_w[64,128], centroids[64,128]
#define ND 64
#define DD 128
#define KK 64
#define LL 3136             // 56*56
#define SPLIT 14            // L-chunks per sample
#define LCHUNK (LL/SPLIT)   // 224
#define ITERS (LCHUNK/32)   // 7 tiles of 32 pixels

typedef __attribute__((ext_vector_type(16))) __bf16 v16bf;
typedef __attribute__((ext_vector_type(8)))  float  v8f;

union Frag { v16bf v; unsigned int u[8]; };

// Native RNE f32 -> bf16 (clang emits the gfx1250 bf16 convert path)
__device__ __forceinline__ unsigned short f2bf(float f) {
    __bf16 h = (__bf16)f;
    return __builtin_bit_cast(unsigned short, h);
}
__device__ __forceinline__ float bf2f(unsigned short h) {
    return __uint_as_float(((unsigned int)h) << 16);
}

// ---------------------------------------------------------------------------
// Main kernel: normalize -> logits (WMMA bf16) -> softmax -> VLAD agg (WMMA)
// grid = ND*SPLIT blocks, 128 threads (4 waves); wave w owns cluster tile w*16..
// Next x-tile is warmed into cache with global_prefetch_b8 (no VGPR cost)
// while the GEMM/softmax phases of the current tile run.
// ---------------------------------------------------------------------------
__global__ __launch_bounds__(128) void nv_main(const float* __restrict__ x,
                                               const float* __restrict__ conv_w,
                                               float* __restrict__ vlad_ws,
                                               float* __restrict__ asum_ws)
{
    __shared__ __align__(16) unsigned short xbf[32 * DD];   // [pixel][chan] bf16
    __shared__ __align__(16) unsigned short xbfT[DD * 32];  // [chan][pixel] bf16
    __shared__ __align__(16) unsigned short a_lds[KK * 32]; // [cluster][pixel] bf16
    __shared__ float logits_lds[32 * KK];                   // [pixel][cluster] f32
    __shared__ float ps4[4 * 32];
    __shared__ float pm[4 * 32];
    __shared__ float psum[4 * 32];

    const int t    = threadIdx.x;
    const int lane = t & 31;
    const int w    = t >> 5;        // wave id 0..3
    const int half = lane >> 4;     // lane-group within wave
    const int l15  = lane & 15;
    const int n     = blockIdx.x / SPLIT;
    const int chunk = blockIdx.x % SPLIT;
    const int lbase0 = chunk * LCHUNK;

    const int p = t & 31;           // pixel index for load/softmax phases
    const int q = t >> 5;           // group index (== w)

    // Per-thread base pointer into x for its (pixel, channel-group) column
    const float* xcol = x + ((size_t)n * DD + q * 32) * LL + lbase0 + p;

    // Preload conv_w B-fragments (B[c,k] = w[k,c]) for this wave's 16 clusters
    Frag bwf[4];
    {
        const int k = w * 16 + l15;
        #pragma unroll
        for (int j = 0; j < 4; ++j)
            #pragma unroll
            for (int v = 0; v < 8; ++v) {
                int c = j * 32 + half * 16 + 2 * v;
                unsigned int lo = (unsigned int)f2bf(conv_w[k * DD + c]);
                unsigned int hi = (unsigned int)f2bf(conv_w[k * DD + c + 1]);
                bwf[j].u[v] = lo | (hi << 16);
            }
    }

    // Persistent VLAD accumulators: 8 d-tiles of 16x16 f32
    const v8f vz = {0.f,0.f,0.f,0.f,0.f,0.f,0.f,0.f};
    v8f vacc[8];
    #pragma unroll
    for (int i = 0; i < 8; ++i) vacc[i] = vz;

    float asumreg = 0.f;
    const int ak = t & 63;          // cluster for asum accumulation
    const int ap = (t >> 6) * 16;   // pixel-half base

    // Warm the first tile
    #pragma unroll
    for (int i = 0; i < 32; ++i)
        __builtin_prefetch(xcol + (size_t)i * LL, 0, 1);

    for (int it = 0; it < ITERS; ++it) {
        const float* xp = xcol + it * 32;

        // ---- load 32 pixels x 128 channels, per-pixel L2 normalize ----
        float val[32];
        float ss = 0.f;
        #pragma unroll
        for (int i = 0; i < 32; ++i) { float f = xp[(size_t)i * LL]; val[i] = f; ss += f * f; }
        ps4[q * 32 + p] = ss;
        __syncthreads();
        float tot = ps4[p] + ps4[32 + p] + ps4[64 + p] + ps4[96 + p];
        float inv = 1.f / fmaxf(sqrtf(tot), 1e-12f);
        #pragma unroll
        for (int i = 0; i < 32; ++i) {
            unsigned short h = f2bf(val[i] * inv);
            int d = q * 32 + i;
            xbf[p * DD + d] = h;
            xbfT[d * 32 + p] = h;
        }

        // ---- warm next tile's cachelines (no VGPRs, no counters) ----
        if (it + 1 < ITERS) {
            const float* xnp = xcol + (it + 1) * 32;
            #pragma unroll
            for (int i = 0; i < 32; ++i)
                __builtin_prefetch(xnp + (size_t)i * LL, 0, 1);
        }
        __syncthreads();

        // ---- GEMM-1: logits[p,k] = sum_d xn[p,d]*w[k,d] ----
        #pragma unroll
        for (int m = 0; m < 2; ++m) {
            v8f acc = vz;
            #pragma unroll
            for (int j = 0; j < 4; ++j) {
                Frag a;
                int row = m * 16 + l15;
                #pragma unroll
                for (int v = 0; v < 8; ++v) {
                    int c = j * 32 + half * 16 + v * 2;
                    a.u[v] = *(const unsigned int*)&xbf[row * DD + c];
                }
                acc = __builtin_amdgcn_wmma_f32_16x16x32_bf16(
                        false, a.v, false, bwf[j].v, (short)0, acc, false, false);
            }
            #pragma unroll
            for (int j = 0; j < 8; ++j) {
                int pr = m * 16 + j + half * 8;
                logits_lds[pr * KK + w * 16 + l15] = acc[j];
            }
        }
        __syncthreads();

        // ---- softmax over 64 clusters per pixel (4 groups of 16) ----
        {
            float mx = -3.0e38f;
            #pragma unroll
            for (int k = 0; k < 16; ++k) mx = fmaxf(mx, logits_lds[p * KK + q * 16 + k]);
            pm[q * 32 + p] = mx;
            __syncthreads();
            mx = fmaxf(fmaxf(pm[p], pm[32 + p]), fmaxf(pm[64 + p], pm[96 + p]));
            float s = 0.f;
            #pragma unroll
            for (int k = 0; k < 16; ++k) {
                float e = __expf(logits_lds[p * KK + q * 16 + k] - mx);
                logits_lds[p * KK + q * 16 + k] = e;
                s += e;
            }
            psum[q * 32 + p] = s;
            __syncthreads();
            float sinv = 1.f / (psum[p] + psum[32 + p] + psum[64 + p] + psum[96 + p]);
            #pragma unroll
            for (int k = 0; k < 16; ++k)
                a_lds[(q * 16 + k) * 32 + p] = f2bf(logits_lds[p * KK + q * 16 + k] * sinv);
        }
        __syncthreads();

        // ---- asum[k] partial (register accumulation) ----
        {
            float s = 0.f;
            #pragma unroll
            for (int i = 0; i < 16; ++i) s += bf2f(a_lds[ak * 32 + ap + i]);
            asumreg += s;
        }

        // ---- GEMM-2: vlad[k,d] += sum_p a[p,k]*xn[p,d] ----
        {
            Frag a;
            const int k = w * 16 + l15;
            #pragma unroll
            for (int v = 0; v < 8; ++v) {
                int pp = half * 16 + 2 * v;
                a.u[v] = *(const unsigned int*)&a_lds[k * 32 + pp];
            }
            #pragma unroll
            for (int td = 0; td < 8; ++td) {
                Frag b;
                int c = td * 16 + l15;
                #pragma unroll
                for (int v = 0; v < 8; ++v) {
                    int pp = half * 16 + 2 * v;
                    b.u[v] = *(const unsigned int*)&xbfT[c * 32 + pp];
                }
                vacc[td] = __builtin_amdgcn_wmma_f32_16x16x32_bf16(
                            false, a.v, false, b.v, (short)0, vacc[td], false, false);
            }
        }
        __syncthreads();
    }

    // ---- flush partials (workspace zeroed before launch) ----
    #pragma unroll
    for (int td = 0; td < 8; ++td)
        #pragma unroll
        for (int j = 0; j < 8; ++j) {
            int k  = w * 16 + j + half * 8;
            int ch = td * 16 + l15;
            atomicAdd(&vlad_ws[((size_t)n * KK + k) * DD + ch], vacc[td][j]);
        }
    atomicAdd(&asum_ws[n * KK + ak], asumreg);
}

// ---------------------------------------------------------------------------
// Finalize: vlad -= asum*c, intra-normalize per (n,k), then global L2 per n
// grid = ND blocks, 128 threads (thread = channel d)
// ---------------------------------------------------------------------------
__global__ __launch_bounds__(128) void nv_finalize(const float* __restrict__ vlad_ws,
                                                   const float* __restrict__ asum_ws,
                                                   const float* __restrict__ centroids,
                                                   float* __restrict__ out)
{
    __shared__ float red[128];
    const int n = blockIdx.x;
    const int d = threadIdx.x;
    float gss = 0.f;
    for (int k = 0; k < KK; ++k) {
        float v = vlad_ws[((size_t)n * KK + k) * DD + d]
                - asum_ws[n * KK + k] * centroids[k * DD + d];
        red[d] = v * v;
        __syncthreads();
        for (int s = 64; s > 0; s >>= 1) {
            if (d < s) red[d] += red[d + s];
            __syncthreads();
        }
        float tot = red[0];
        __syncthreads();
        float inv = 1.f / fmaxf(sqrtf(tot), 1e-12f);
        out[(size_t)n * KK * DD + k * DD + d] = v * inv;
        gss += tot * inv * inv;
    }
    float ginv = 1.f / fmaxf(sqrtf(gss), 1e-12f);
    for (int k = 0; k < KK; ++k)
        out[(size_t)n * KK * DD + k * DD + d] *= ginv;
}

extern "C" void kernel_launch(void* const* d_in, const int* in_sizes, int n_in,
                              void* d_out, int out_size, void* d_ws, size_t ws_size,
                              hipStream_t stream) {
    const float* x         = (const float*)d_in[0];
    const float* conv_w    = (const float*)d_in[1];
    const float* centroids = (const float*)d_in[2];
    float* out = (float*)d_out;

    float* vlad_ws = (float*)d_ws;                        // [ND][KK][DD] f32
    float* asum_ws = vlad_ws + (size_t)ND * KK * DD;      // [ND][KK]     f32
    size_t zero_bytes = ((size_t)ND * KK * DD + (size_t)ND * KK) * sizeof(float);

    hipMemsetAsync(d_ws, 0, zero_bytes, stream);
    nv_main<<<ND * SPLIT, 128, 0, stream>>>(x, conv_w, vlad_ws, asum_ws);
    nv_finalize<<<ND, 128, 0, stream>>>(vlad_ws, asum_ws, centroids, out);
}
